// _AA_DeformConv2d_63668595196241
// MI455X (gfx1250) — matrix operand order repaired and checked by reading, not compiled, not run on hardware
//
#include <hip/hip_runtime.h>
#include <hip/hip_bf16.h>
#include <cstdint>

// ---------------- problem constants ----------------
constexpr int B_   = 4;
constexpr int C_   = 128;
constexpr int Cg_  = 64;     // channels per deform group
constexpr int DG_  = 2;
constexpr int K_   = 9;      // 3x3 taps
constexpr int H_   = 96;
constexpr int W_   = 192;
constexpr int Ho_  = 96;
constexpr int Wo_  = 192;
constexpr int Cout_= 128;
constexpr int KK_  = DG_ * K_ * Cg_;   // 1152 = main GEMM K
constexpr int KOFF_= K_ * Cg_;         // 576  = offset GEMM K (per conv group)
constexpr int HW_  = H_ * W_;
constexpr int PIX  = Ho_ * Wo_;        // 18432
constexpr int NPIX = B_ * PIX;         // 73728 = GEMM N
constexpr int OFF_CH = DG_ * 3 * K_;   // 54

// workspace layout (bytes)
constexpr size_t WS_WBF  = 0;                                   // 128*1152*2 = 294912
constexpr size_t WS_PY   = 294912;
constexpr size_t WS_PX   = WS_PY + (size_t)B_*DG_*K_*PIX*4;     // 5,308,416 each
constexpr size_t WS_MK   = WS_PX + (size_t)B_*DG_*K_*PIX*4;
constexpr size_t WS_WOFF = WS_MK + (size_t)B_*DG_*K_*PIX*4;     // 2*32*576*2 = 73728

typedef __attribute__((ext_vector_type(16))) __bf16 v16bf;
typedef __attribute__((ext_vector_type(8)))  float  v8f;

__device__ __forceinline__ unsigned short f2bf(float f) {
  union { float f; unsigned int u; } v; v.f = f;
  unsigned int u = v.u;
  return (unsigned short)((u + 0x7FFFu + ((u >> 16) & 1u)) >> 16);  // RNE
}

// A 16x32 bf16 fragment: VGPR r holds K = (r/4)*16 + half*8 + (r%4)*2 (+1)
__device__ __forceinline__ int ka_of(int r, int half) {
  return ((r >> 2) << 4) + (half << 3) + ((r & 3) << 1);
}
// B 32x16 bf16 fragment: VGPR r holds K = (r/4)*16 + half*8 + (r%4)*2 (+1)
__device__ __forceinline__ int kb_of(int r, int half) {
  return ((r >> 2) << 4) + (half << 3) + ((r & 3) << 1);
}

// ---------------- pack main weights: w_deform -> bf16 Wbf[o][kk] ----------------
// kk = (g*9 + k)*64 + c ; w_deform is (Cout, C, 3, 3)
__global__ __launch_bounds__(256) void pack_w_kernel(
    const float* __restrict__ wd, unsigned short* __restrict__ Wbf) {
  int idx = blockIdx.x * 256 + threadIdx.x;
  if (idx >= Cout_ * KK_) return;
  int o  = idx / KK_;
  int kk = idx - o * KK_;
  int g  = kk / KOFF_;
  int r  = kk - g * KOFF_;
  int k  = r / Cg_;
  int c  = r - k * Cg_;
  Wbf[idx] = f2bf(wd[(o * C_ + g * Cg_ + c) * K_ + k]);
}

// ------------ pack offset weights: w_offset -> bf16 Woffbf[g*32+ocl][kk2] -------
// kk2 = k*64 + c ; rows ocl>=27 are zero padding. w_offset is (54, 64, 3, 3)
__global__ __launch_bounds__(256) void pack_woff_kernel(
    const float* __restrict__ wo, unsigned short* __restrict__ Woffbf) {
  int idx = blockIdx.x * 256 + threadIdx.x;
  if (idx >= 2 * 32 * KOFF_) return;
  int row = idx / KOFF_;
  int kk  = idx - row * KOFF_;
  int g   = row >> 5;
  int ocl = row & 31;
  int k   = kk >> 6;
  int c   = kk & 63;
  float v = (ocl < 27) ? wo[((g * 27 + ocl) * Cg_ + c) * K_ + k] : 0.f;
  Woffbf[idx] = f2bf(v);
}

// -------- offset conv as WMMA GEMM (dil=2, pad=2, groups=2) + fused transform ---
// Block = 256 thr = 8 waves; N-tile = 32 pixels. M: 4 tiles of 16 (2 per group).
// wave w: mtile = w&3 (group g = mtile>>1), nsub = w>>2. 18 K-chunks of 32.
__global__ __launch_bounds__(256) void offset_wmma_kernel(
    const float* __restrict__ x, const unsigned short* __restrict__ Woffbf,
    const float* __restrict__ b_offset, float* __restrict__ pyA,
    float* __restrict__ pxA, float* __restrict__ mkA) {
  constexpr int VSTO = 34;                         // bf16 row stride
  __shared__ int   s_ho[32];
  __shared__ int   s_wo[32];
  __shared__ int   s_lin2[32];                     // clamped source index
  __shared__ float s_ok[32];                       // 1.0 valid / 0.0 OOB
  __shared__ __attribute__((aligned(16))) unsigned short s_b2[2][32][VSTO];

  const int tid  = threadIdx.x;
  const int wave = tid >> 5;
  const int lane = tid & 31;
  const int col  = lane & 15;
  const int half = lane >> 4;
  const int mt   = wave & 3;
  const int g    = mt >> 1;
  const int nsub = wave >> 2;
  const int n0   = blockIdx.x * 32;
  const int bimg = n0 / PIX;                       // PIX % 32 == 0 -> uniform b

  if (tid < 32) {                                  // per-pixel (ho,wo) once
    int hw = (n0 + tid) - bimg * PIX;
    s_ho[tid] = hw / Wo_;
    s_wo[tid] = hw - (hw / Wo_) * Wo_;
  }

  v8f acc = {};

  for (int ch = 0; ch < 18; ++ch) {
    const int k  = ch >> 1;
    const int c0 = (ch & 1) << 5;
    __syncthreads();
    if (tid < 32) {                                // im2col source per pixel
      int iy = s_ho[tid] - 2 + (k / 3) * 2;
      int ix = s_wo[tid] - 2 + (k % 3) * 2;
      bool ok = (iy >= 0) && (iy < H_) && (ix >= 0) && (ix < W_);
      int yc = min(max(iy, 0), H_ - 1);
      int xc = min(max(ix, 0), W_ - 1);
      s_lin2[tid] = yc * W_ + xc;                  // always in-bounds
      s_ok[tid]   = ok ? 1.f : 0.f;
    }
    __syncthreads();
    #pragma unroll                                  // branchless fill: 2*32*32
    for (int j = 0; j < 8; ++j) {
      int e  = tid + j * 256;
      int n  = e & 31;
      int cc = (e >> 5) & 31;
      int gg = e >> 10;
      float v = x[(size_t)(bimg * C_ + gg * Cg_ + c0 + cc) * HW_ + s_lin2[n]]
                * s_ok[n];
      s_b2[gg][n][cc] = f2bf(v);
    }
    __syncthreads();
    const unsigned short* wrow =
        Woffbf + (size_t)((g << 5) + ((mt & 1) << 4) + col) * KOFF_ + k * Cg_ + c0;
    union { v16bf v; unsigned int u[8]; } a, bm;
    #pragma unroll
    for (int r = 0; r < 8; ++r) {
      a.u[r]  = *(const unsigned int*)(wrow + ka_of(r, half));
      bm.u[r] = *(const unsigned int*)(&s_b2[g][(nsub << 4) + col][kb_of(r, half)]);
    }
    acc = __builtin_amdgcn_wmma_f32_16x16x32_bf16(
        false, a.v, false, bm.v, (short)0, acc, false, false);
  }

  // epilogue: bias + offset/mask transform (C/D: lane = col; VGPR r -> M = half*8+r)
  int hw = (n0 + (nsub << 4) + col) - bimg * PIX;
  int ho = hw / Wo_, wo = hw - ho * Wo_;
  #pragma unroll
  for (int r = 0; r < 8; ++r) {
    int ocl = ((mt & 1) << 4) + (half << 3) + r;
    if (ocl >= 27) continue;                        // padded rows
    int ocg = g * 27 + ocl;                         // global conv channel 0..53
    float v = acc[r] + b_offset[ocg];
    if (ocg < DG_ * 2 * K_) {                       // offset: ocg = (gd*9+kd)*2+d
      int d  = ocg & 1;
      int gk = ocg >> 1;
      int gd = gk / K_, kd = gk % K_;
      int oidx = ((bimg * DG_ + gd) * K_ + kd) * PIX + hw;
      if (d == 0) pyA[oidx] = (float)(ho - 2 + (kd / 3) * 2) + v;
      else        pxA[oidx] = (float)(wo - 2 + (kd % 3) * 2) + v;
    } else {                                        // mask
      int gk = ocg - DG_ * 2 * K_;
      int gd = gk / K_, kd = gk % K_;
      mkA[((bimg * DG_ + gd) * K_ + kd) * PIX + hw] = 2.f / (1.f + __expf(-v));
    }
  }
}

// ---------------- fused bilinear-gather + bf16 WMMA main GEMM -------------------
// Block = 256 thr = 8 waves; N-tile = 32 pixels; wave w owns rows [16w,16w+16)
// and both 16-pixel N-subtiles (2 accumulators, 4 WMMAs per (g,k), A reused).
__global__ __launch_bounds__(256) void dcn_wmma_kernel(
    const float* __restrict__ x, const unsigned short* __restrict__ Wbf,
    const float* __restrict__ pyA, const float* __restrict__ pxA,
    const float* __restrict__ mkA, float* __restrict__ out) {
  constexpr int VST = 68;                           // bf16 row stride (64 ch + pad)
  __shared__ int   s_lin[4][32];
  __shared__ float s_wt[4][32];
  __shared__ float s_m[32];
  __shared__ __attribute__((aligned(16))) unsigned short s_val[32][VST];

  const int tid  = threadIdx.x;
  const int wave = tid >> 5;
  const int lane = tid & 31;
  const int col  = lane & 15;
  const int half = lane >> 4;
  const int n0   = blockIdx.x * 32;
  const int bimg = n0 / PIX;                        // PIX % 32 == 0 -> uniform b
  const int xbase = bimg * C_ * HW_;

  v8f acc0 = {}, acc1 = {};

  for (int g = 0; g < DG_; ++g) {
    for (int kk = 0; kk < K_; ++kk) {
      __syncthreads();                              // previous tile fully consumed
      if (tid < 32) {                               // per-pixel taps/weights/mask
        int hw = (n0 + tid) - bimg * PIX;
        int oidx = ((bimg * DG_ + g) * K_ + kk) * PIX + hw;
        float py = pyA[oidx];
        float px = pxA[oidx];
        s_m[tid] = mkA[oidx];
        float y0f = floorf(py), x0f = floorf(px);
        float ty = py - y0f, tx = px - x0f;
        int y0 = (int)y0f, x0 = (int)x0f;
        float wy[2] = {1.f - ty, ty};
        float wx[2] = {1.f - tx, tx};
        #pragma unroll
        for (int t = 0; t < 4; ++t) {
          int dy = t >> 1, dx = t & 1;
          int yy = y0 + dy, xx = x0 + dx;
          bool ok = (yy >= 0) && (yy < H_) && (xx >= 0) && (xx < W_);
          int yc = min(max(yy, 0), H_ - 1);
          int xc = min(max(xx, 0), W_ - 1);
          s_lin[t][tid] = yc * W_ + xc;
          s_wt[t][tid]  = ok ? wy[dy] * wx[dx] : 0.f;
        }
      }
      __syncthreads();
      // gather-blend 64 channels x 32 pixels into bf16 tile (branchless)
      #pragma unroll
      for (int j = 0; j < 8; ++j) {
        int e = tid + j * 256;
        int c = e >> 5;
        int i = e & 31;
        const float* xp = x + (xbase + (g * Cg_ + c) * HW_);
        float v = s_wt[0][i] * xp[s_lin[0][i]]
                + s_wt[1][i] * xp[s_lin[1][i]]
                + s_wt[2][i] * xp[s_lin[2][i]]
                + s_wt[3][i] * xp[s_lin[3][i]];
        s_val[i][c] = f2bf(v * s_m[i]);
      }
      __syncthreads();
      // two K=32 chunks cover c = 0..63; A fragment shared by both N-subtiles
      const unsigned short* wrow =
          Wbf + (size_t)((wave << 4) + col) * KK_ + (g * K_ + kk) * Cg_;
      #pragma unroll
      for (int c0 = 0; c0 < Cg_; c0 += 32) {
        union { v16bf v; unsigned int u[8]; } a, b0, b1;
        #pragma unroll
        for (int r = 0; r < 8; ++r) {
          a.u[r]  = *(const unsigned int*)(wrow + c0 + ka_of(r, half));
          int kb  = kb_of(r, half);
          b0.u[r] = *(const unsigned int*)(&s_val[col][c0 + kb]);
          b1.u[r] = *(const unsigned int*)(&s_val[16 + col][c0 + kb]);
        }
        acc0 = __builtin_amdgcn_wmma_f32_16x16x32_bf16(
            false, a.v, false, b0.v, (short)0, acc0, false, false);
        acc1 = __builtin_amdgcn_wmma_f32_16x16x32_bf16(
            false, a.v, false, b1.v, (short)0, acc1, false, false);
      }
    }
  }

  // C/D layout: lane = N col; VGPR r -> M = half*8 + r
  int hw0 = (n0 + col) - bimg * PIX;
  size_t obase = (size_t)(bimg * Cout_) * PIX;
  #pragma unroll
  for (int r = 0; r < 8; ++r) {
    int o = (wave << 4) + (half << 3) + r;
    out[obase + (size_t)o * PIX + hw0]      = acc0[r];
    out[obase + (size_t)o * PIX + hw0 + 16] = acc1[r];
  }
}

// ---------------- launcher ----------------
extern "C" void kernel_launch(void* const* d_in, const int* in_sizes, int n_in,
                              void* d_out, int out_size, void* d_ws, size_t ws_size,
                              hipStream_t stream) {
  const float* x        = (const float*)d_in[0];
  const float* w_offset = (const float*)d_in[1];
  const float* b_offset = (const float*)d_in[2];
  const float* w_deform = (const float*)d_in[3];
  float* out = (float*)d_out;

  char* ws = (char*)d_ws;
  unsigned short* Wbf    = (unsigned short*)(ws + WS_WBF);
  float* pyA             = (float*)(ws + WS_PY);
  float* pxA             = (float*)(ws + WS_PX);
  float* mkA             = (float*)(ws + WS_MK);
  unsigned short* Woffbf = (unsigned short*)(ws + WS_WOFF);

  pack_w_kernel<<<(Cout_ * KK_ + 255) / 256, 256, 0, stream>>>(w_deform, Wbf);
  pack_woff_kernel<<<(2 * 32 * KOFF_ + 255) / 256, 256, 0, stream>>>(w_offset, Woffbf);
  offset_wmma_kernel<<<NPIX / 32, 256, 0, stream>>>(
      x, Woffbf, b_offset, pyA, pxA, mkA);
  dcn_wmma_kernel<<<NPIX / 32, 256, 0, stream>>>(x, Wbf, pyA, pxA, mkA, out);
}
